// SemanticGatheringScatteringTransformer_55070070669425
// MI455X (gfx1250) — matrix operand order repaired and checked by reading, not compiled, not run on hardware
//
#include <hip/hip_runtime.h>
#include <hip/hip_bf16.h>
#include <math.h>

// ---------------- sizes ----------------
#define BSZ   8
#define CCH   256
#define NPIX  4096            // H*W
#define NHD   8
#define DH    32
#define FFD   1024
#define NLAY  3
#define TOK   64              // K pooled tokens
#define MROWS (BSZ*NPIX)      // 32768
#define KVROWS (BSZ*TOK)      // 512
#define ATT_SCALE 0.17677669529663687f   // 1/sqrt(32)

typedef _Float16 half_t;
typedef __attribute__((ext_vector_type(16))) _Float16 v16h;
typedef __attribute__((ext_vector_type(8)))  float    v8f;
typedef __attribute__((ext_vector_type(2)))  _Float16 hpair;

// ---- CDNA5 async global->LDS (ASYNCcnt path), portable inline asm ----
__device__ __forceinline__ void async_load_b128(const void* gptr, void* lptr) {
    uint32_t loff = (uint32_t)(uintptr_t)(__attribute__((address_space(3))) const void*)lptr;
    uint64_t ga   = (uint64_t)(uintptr_t)gptr;
    asm volatile("global_load_async_to_lds_b128 %0, %1, off"
                 :: "v"(loff), "v"(ga) : "memory");
}
__device__ __forceinline__ void wait_async0() {
    asm volatile("s_wait_asynccnt 0x0" ::: "memory");
}

// =====================================================================
// Generic f16 WMMA GEMM:  C[M,Nd] = A[M,Kd] * W[Nd,Kd]^T + bias
// block = 256 thr (8 waves). Block tile 128x64, wave tile 32x32.
// A tile staged via GLOBAL_LOAD_ASYNC_TO_LDS_B128 (no VGPR round-trip);
// W tile staged via VGPRs (needs transpose into Bs[k][n]).
// =====================================================================
#define BM 128
#define BN 64
#define KS 32
#define ASTR (KS+8)
#define BSTR (BN+8)

__global__ __launch_bounds__(256)
void gemm_f16_wmma(const half_t* __restrict__ A, const half_t* __restrict__ W,
                   const float* __restrict__ bias, float* __restrict__ Cf,
                   half_t* __restrict__ Ch, int M, int Nd, int Kd, int relu)
{
    __shared__ half_t As[BM * ASTR];
    __shared__ half_t Bs[KS * BSTR];   // transposed: Bs[k][n]

    const int tid  = threadIdx.x;
    const int lane = tid & 31;
    const int wave = tid >> 5;
    const int wm = (wave & 3) * 32;    // 4 waves along M
    const int wn = (wave >> 2) * 32;   // 2 waves along N
    const int bm = blockIdx.y * BM;
    const int bn = blockIdx.x * BN;
    const int hs = lane >> 4, lr = lane & 15;

    v8f acc[2][2] = {};

    for (int k0 = 0; k0 < Kd; k0 += KS) {
        // stage A tile: 128 rows x 32 halfs = 512 x b128, async direct to LDS
        #pragma unroll
        for (int i = 0; i < 2; ++i) {
            int idx = tid + i * 256;
            int r   = idx >> 2;
            int c8  = (idx & 3) * 8;
            async_load_b128(A + (size_t)(bm + r) * Kd + k0 + c8,
                            &As[r * ASTR + c8]);
        }
        // stage W tile transposed into Bs[k][n] (VGPR path, transpose)
        {
            int r  = tid >> 2;              // n row 0..63
            int c8 = (tid & 3) * 8;
            uint4 v = *(const uint4*)(W + (size_t)(bn + r) * Kd + k0 + c8);
            const half_t* hv = (const half_t*)&v;
            #pragma unroll
            for (int j = 0; j < 8; ++j) Bs[(c8 + j) * BSTR + r] = hv[j];
        }
        if (k0 + KS < Kd) {  // prefetch next W panel -> global_prefetch_b8
            __builtin_prefetch(W + (size_t)(bn + (tid & 63)) * Kd + k0 + KS, 0, 1);
        }
        wait_async0();            // our async A-tile writes to LDS are done
        __syncthreads();          // + compiler-inserted dscnt wait for Bs stores

        // A fragments (16x32 f16 layout, ISA 7.12.2)
        v16h afrag[2], bfrag[2];
        #pragma unroll
        for (int s = 0; s < 2; ++s) {
            const half_t* ar = &As[(wm + 16 * s + lr) * ASTR];
            #pragma unroll
            for (int p = 0; p < 8; ++p) {
                int kk = (p < 4) ? (8 * hs + 2 * p) : (8 + 8 * hs + 2 * p);
                hpair t = *(const hpair*)(ar + kk);
                afrag[s][2 * p]     = t.x;
                afrag[s][2 * p + 1] = t.y;
            }
        }
        // B fragments: lane holds N column lr, elements sweep K (same order as A)
        #pragma unroll
        for (int s = 0; s < 2; ++s) {
            const half_t* br = &Bs[wn + 16 * s + lr];
            #pragma unroll
            for (int p = 0; p < 8; ++p) {
                int kk = (p < 4) ? (8 * hs + 2 * p) : (8 + 8 * hs + 2 * p);
                bfrag[s][2 * p]     = br[kk * BSTR];
                bfrag[s][2 * p + 1] = br[(kk + 1) * BSTR];
            }
        }
        #pragma unroll
        for (int i = 0; i < 2; ++i)
            #pragma unroll
            for (int j = 0; j < 2; ++j)
                acc[i][j] = __builtin_amdgcn_wmma_f32_16x16x32_f16(
                    false, afrag[i], false, bfrag[j], (short)0, acc[i][j], false, false);
        __syncthreads();
    }

    // epilogue (C/D layout: col = lane&15, row = r + 8*(lane>>4))
    #pragma unroll
    for (int i = 0; i < 2; ++i)
        #pragma unroll
        for (int j = 0; j < 2; ++j) {
            int col = bn + wn + 16 * j + lr;
            float bv = bias ? bias[col] : 0.f;
            #pragma unroll
            for (int r = 0; r < 8; ++r) {
                int row = bm + wm + 16 * i + r + 8 * hs;
                float v = acc[i][j][r] + bv;
                if (relu) v = v > 0.f ? v : 0.f;
                size_t off = (size_t)row * Nd + col;
                if (Cf) Cf[off] = v;
                if (Ch) Ch[off] = (half_t)v;
            }
        }
}

// =====================================================================
// Fused attention: per (b,h): scores=QK^T*s, softmax(64), out=att*V.
// block = 256 thr (8 waves x 16 query rows = 128 rows/block)
// =====================================================================
__global__ __launch_bounds__(256)
void attn_wmma(const half_t* __restrict__ qq, const half_t* __restrict__ kk,
               const half_t* __restrict__ vv, half_t* __restrict__ o)
{
    __shared__ half_t KsT[DH * 72];          // [d][tok] transposed
    __shared__ half_t Vs [TOK * 40];         // [tok][d]
    __shared__ float  Ssc[8 * 16 * 68];      // per-wave scores
    __shared__ half_t Att[8 * 16 * 72];      // per-wave normalized att (f16)

    const int tid  = threadIdx.x;
    const int lane = tid & 31;
    const int wave = tid >> 5;
    const int hs = lane >> 4, lr = lane & 15;
    const int bh = blockIdx.y;
    const int b  = bh / NHD, h = bh % NHD;
    const int rowbase = blockIdx.x * 128 + wave * 16;   // row within image

    // stage K (transposed, VGPR path) and V (async direct to LDS)
    {
        int t  = tid >> 2;
        int c8 = (tid & 3) * 8;
        uint4 kv4 = *(const uint4*)(kk + ((size_t)(b * TOK + t)) * CCH + h * DH + c8);
        const half_t* hv = (const half_t*)&kv4;
        #pragma unroll
        for (int j = 0; j < 8; ++j) KsT[(c8 + j) * 72 + t] = hv[j];
        async_load_b128(vv + ((size_t)(b * TOK + t)) * CCH + h * DH + c8,
                        &Vs[t * 40 + c8]);
    }
    wait_async0();
    __syncthreads();

    // Q fragment from global (A 16x32 layout; k-dim = head dim 32)
    v16h aq;
    {
        const half_t* qr = qq + ((size_t)(b * NPIX + rowbase + lr)) * CCH + h * DH;
        #pragma unroll
        for (int p = 0; p < 8; ++p) {
            int kd = (p < 4) ? (8 * hs + 2 * p) : (8 + 8 * hs + 2 * p);
            hpair t = *(const hpair*)(qr + kd);
            aq[2 * p] = t.x; aq[2 * p + 1] = t.y;
        }
    }
    // scores: 4 token-tiles of 16, single k-step (d=32)
    v8f sc[4];
    {
        #pragma unroll
        for (int t4 = 0; t4 < 4; ++t4) {
            v16h bk;
            const half_t* kr = &KsT[16 * t4 + lr];
            #pragma unroll
            for (int p = 0; p < 8; ++p) {
                int d = (p < 4) ? (8 * hs + 2 * p) : (8 + 8 * hs + 2 * p);
                bk[2 * p] = kr[d * 72]; bk[2 * p + 1] = kr[(d + 1) * 72];
            }
            v8f z = {};
            sc[t4] = __builtin_amdgcn_wmma_f32_16x16x32_f16(false, aq, false, bk,
                                                            (short)0, z, false, false);
        }
    }
    // dump scaled scores to LDS
    #pragma unroll
    for (int t4 = 0; t4 < 4; ++t4)
        #pragma unroll
        for (int r = 0; r < 8; ++r)
            Ssc[(wave * 16 + r + 8 * hs) * 68 + 16 * t4 + lr] = sc[t4][r] * ATT_SCALE;
    __syncthreads();

    // softmax over 64: lane lr handles row lr, half hs covers 32 cols
    {
        float* srow = &Ssc[(wave * 16 + lr) * 68 + 32 * hs];
        float m = -1e30f;
        #pragma unroll 8
        for (int j = 0; j < 32; ++j) m = fmaxf(m, srow[j]);
        m = fmaxf(m, __shfl_xor(m, 16, 32));
        float s = 0.f;
        #pragma unroll 8
        for (int j = 0; j < 32; ++j) { float e = __expf(srow[j] - m); srow[j] = e; s += e; }
        s += __shfl_xor(s, 16, 32);
        float inv = 1.f / s;
        half_t* arow = &Att[(wave * 16 + lr) * 72 + 32 * hs];
        #pragma unroll 8
        for (int j = 0; j < 32; ++j) arow[j] = (half_t)(srow[j] * inv);
    }
    __syncthreads();

    // out = att(16x64) * V(64x32): 2 k-steps, 2 n-tiles
    v8f oacc[2] = {};
    #pragma unroll
    for (int ks = 0; ks < 2; ++ks) {
        v16h aa;
        const half_t* ar = &Att[(wave * 16 + lr) * 72 + 32 * ks];
        #pragma unroll
        for (int p = 0; p < 8; ++p) {
            int kt = (p < 4) ? (8 * hs + 2 * p) : (8 + 8 * hs + 2 * p);
            hpair t = *(const hpair*)(ar + kt);
            aa[2 * p] = t.x; aa[2 * p + 1] = t.y;
        }
        #pragma unroll
        for (int nt = 0; nt < 2; ++nt) {
            v16h bv;
            const half_t* vr = &Vs[(32 * ks) * 40 + 16 * nt + lr];
            #pragma unroll
            for (int p = 0; p < 8; ++p) {
                int kt = (p < 4) ? (8 * hs + 2 * p) : (8 + 8 * hs + 2 * p);
                bv[2 * p] = vr[kt * 40]; bv[2 * p + 1] = vr[(kt + 1) * 40];
            }
            oacc[nt] = __builtin_amdgcn_wmma_f32_16x16x32_f16(false, aa, false, bv,
                                                              (short)0, oacc[nt], false, false);
        }
    }
    #pragma unroll
    for (int nt = 0; nt < 2; ++nt)
        #pragma unroll
        for (int r = 0; r < 8; ++r)
            o[((size_t)(b * NPIX + rowbase + r + 8 * hs)) * CCH + h * DH + 16 * nt + lr] =
                (half_t)oacc[nt][r];
}

// =====================================================================
// Support kernels (fp32 VALU)
// =====================================================================
__global__ void cvt_f32_f16(const float* __restrict__ src, half_t* __restrict__ dst, int n) {
    int i = blockIdx.x * blockDim.x + threadIdx.x;
    if (i < n) dst[i] = (half_t)src[i];
}

__global__ void zero_i32(int* p, int n) {
    int i = threadIdx.x; if (i < n) p[i] = 0;
}

// concat(x,y) [B,2C,N] -> f16 [B,N,512] with LDS tile transpose
__global__ __launch_bounds__(256)
void tconcat(const float* __restrict__ x, const float* __restrict__ y,
             half_t* __restrict__ dst)
{
    __shared__ float t[32][33];
    int n0 = blockIdx.x * 32, c0 = blockIdx.y * 32, b = blockIdx.z;
    const float* src = (c0 < CCH) ? x + ((size_t)b * CCH + c0) * NPIX
                                  : y + ((size_t)b * CCH + (c0 - CCH)) * NPIX;
    #pragma unroll
    for (int i = 0; i < 4; ++i) {
        int cc = threadIdx.y + 8 * i;
        t[cc][threadIdx.x] = src[(size_t)cc * NPIX + n0 + threadIdx.x];
    }
    __syncthreads();
    #pragma unroll
    for (int i = 0; i < 4; ++i) {
        int nn = threadIdx.y + 8 * i;
        dst[((size_t)b * NPIX + n0 + nn) * 512 + c0 + threadIdx.x] = (half_t)t[threadIdx.x][nn];
    }
}

// saliency dot + sigmoid + per-batch fg/bg counts
__global__ __launch_bounds__(256)
void sal_heat(const float* __restrict__ feat, const float* __restrict__ sw,
              const float* __restrict__ sb, float* __restrict__ heat, int* __restrict__ counts)
{
    __shared__ float red[256];
    int row = blockIdx.x, c = threadIdx.x;
    red[c] = feat[(size_t)row * CCH + c] * sw[c];
    __syncthreads();
    for (int s = 128; s > 0; s >>= 1) { if (c < s) red[c] += red[c + s]; __syncthreads(); }
    if (c == 0) {
        float hv = 1.f / (1.f + __expf(-(red[0] + sb[0])));
        heat[row] = hv;
        int b = row >> 12;
        if (hv >= 0.5f) atomicAdd(&counts[b * 2 + 0], 1);
        else            atomicAdd(&counts[b * 2 + 1], 1);
    }
}

// default output: out[b][c][n] = feat_t[b][n][c]
__global__ void out_default(const float* __restrict__ feat_t, float* __restrict__ out) {
    size_t i = (size_t)blockIdx.x * blockDim.x + threadIdx.x;
    if (i >= (size_t)BSZ * CCH * NPIX) return;
    int n = i & (NPIX - 1); int c = (i >> 12) & (CCH - 1); int b = i >> 20;
    out[i] = feat_t[(((size_t)b << 12) + n) * CCH + c];
}

// wf = feat * hm  (f16)
__global__ void wf_mul(const float* __restrict__ feat, const float* __restrict__ heat,
                       half_t* __restrict__ wf, int side) {
    size_t i = (size_t)blockIdx.x * blockDim.x + threadIdx.x;
    if (i >= (size_t)MROWS * CCH) return;
    float hv = heat[i >> 8];
    if (side) hv = 1.f - hv;
    wf[i] = (half_t)(feat[i] * hv);
}

// q init copy (f32 + f16 mirror)
__global__ void q_init(const float* __restrict__ src, float* __restrict__ q,
                       half_t* __restrict__ qh) {
    size_t i = (size_t)blockIdx.x * blockDim.x + threadIdx.x;
    if (i >= (size_t)MROWS * CCH) return;
    float v = src[i]; q[i] = v; qh[i] = (half_t)v;
}

// pool[b,k,c] = sum_n sw[b,n,k] * wf[b,n,c]    (block per (b,k))
__global__ __launch_bounds__(256)
void pool_red(const float* __restrict__ sw, const half_t* __restrict__ wf,
              float* __restrict__ pool, half_t* __restrict__ poolh)
{
    int bk = blockIdx.x;           // b*64 + k
    int b = bk >> 6, k = bk & 63;
    int c = threadIdx.x;
    float acc = 0.f;
    for (int n = 0; n < NPIX; ++n) {
        size_t row = ((size_t)b << 12) + n;
        float s = sw[row * TOK + k];
        acc += s * (float)wf[row * CCH + c];
    }
    pool[(size_t)bk * CCH + c]  = acc;
    poolh[(size_t)bk * CCH + c] = (half_t)acc;
}

// q = LN(q + delta) * g + b ; writes f32 + f16
__global__ __launch_bounds__(256)
void ln_add(const float* __restrict__ base, const float* __restrict__ delta,
            const float* __restrict__ g, const float* __restrict__ bt,
            float* __restrict__ q, half_t* __restrict__ qh)
{
    __shared__ float red[256];
    int row = blockIdx.x, c = threadIdx.x;
    size_t off = (size_t)row * CCH + c;
    float v = base[off] + delta[off];
    red[c] = v; __syncthreads();
    for (int s = 128; s > 0; s >>= 1) { if (c < s) red[c] += red[c + s]; __syncthreads(); }
    float mean = red[0] * (1.f / CCH);
    __syncthreads();
    float d = v - mean;
    red[c] = d * d; __syncthreads();
    for (int s = 128; s > 0; s >>= 1) { if (c < s) red[c] += red[c + s]; __syncthreads(); }
    float var = red[0] * (1.f / CCH);
    float nv = d * rsqrtf(var + 1e-5f) * g[c] + bt[c];
    q[off] = nv; qh[off] = (half_t)nv;
}

// masked scatter of qs into out (only if batch valid)
__global__ __launch_bounds__(256)
void scatter_side(const float* __restrict__ q, const float* __restrict__ heat,
                  const int* __restrict__ counts, float* __restrict__ out, int side)
{
    int row = blockIdx.x;            // b*N + n
    int b = row >> 12, n = row & (NPIX - 1);
    if (counts[b * 2 + 0] == 0 || counts[b * 2 + 1] == 0) return;
    float hv = heat[row];
    bool sel = (side == 0) ? (hv >= 0.5f) : (hv < 0.5f);
    if (!sel) return;
    int c = threadIdx.x;
    out[(((size_t)b * CCH) + c) * NPIX + n] = q[(size_t)row * CCH + c];
}

// =====================================================================
// Host orchestration
// =====================================================================
static inline void gemm(hipStream_t s, const half_t* A, const half_t* W, const float* bias,
                        float* Cf, half_t* Ch, int M, int Nd, int Kd, int relu) {
    dim3 g(Nd / BN, M / BM), b(256);
    gemm_f16_wmma<<<g, b, 0, s>>>(A, W, bias, Cf, Ch, M, Nd, Kd, relu);
}

extern "C" void kernel_launch(void* const* d_in, const int* in_sizes, int n_in,
                              void* d_out, int out_size, void* d_ws, size_t ws_size,
                              hipStream_t stream) {
    (void)in_sizes; (void)n_in; (void)out_size; (void)ws_size;
    const float* x        = (const float*)d_in[0];
    const float* y        = (const float*)d_in[1];
    const float* proj_w   = (const float*)d_in[2];
    const float* proj_b   = (const float*)d_in[3];
    const float* sal_w    = (const float*)d_in[4];
    const float* sal_b    = (const float*)d_in[5];
    const float* soft_w   = (const float*)d_in[6];
    const float* soft_b   = (const float*)d_in[7];
    const float* ain_w    = (const float*)d_in[8];
    const float* ain_b    = (const float*)d_in[9];
    const float* aout_w   = (const float*)d_in[10];
    const float* aout_b   = (const float*)d_in[11];
    const float* n1_g     = (const float*)d_in[12];
    const float* n1_b     = (const float*)d_in[13];
    const float* f1_w     = (const float*)d_in[14];
    const float* f1_b     = (const float*)d_in[15];
    const float* f2_w     = (const float*)d_in[16];
    const float* f2_b     = (const float*)d_in[17];
    const float* n2_g     = (const float*)d_in[18];
    const float* n2_b     = (const float*)d_in[19];
    float* out = (float*)d_out;

    // ---- workspace carve-up ----
    char* p = (char*)d_ws;
    auto carve = [&](size_t bytes) -> void* {
        void* r = (void*)p; p += (bytes + 255) & ~(size_t)255; return r;
    };
    half_t* w_proj_h = (half_t*)carve((size_t)CCH * 512 * 2);
    half_t* w_soft_h = (half_t*)carve((size_t)2 * TOK * CCH * 2);
    half_t* w_ain_h  = (half_t*)carve((size_t)2 * NLAY * 3 * CCH * CCH * 2);
    half_t* w_aout_h = (half_t*)carve((size_t)2 * NLAY * CCH * CCH * 2);
    half_t* w_f1_h   = (half_t*)carve((size_t)2 * NLAY * FFD * CCH * 2);
    half_t* w_f2_h   = (half_t*)carve((size_t)2 * NLAY * CCH * FFD * 2);
    half_t* xcT_h    = (half_t*)carve((size_t)MROWS * 512 * 2);
    float*  feat_t   = (float*) carve((size_t)MROWS * CCH * 4);
    half_t* feat_h   = (half_t*)carve((size_t)MROWS * CCH * 2);
    float*  heat     = (float*) carve((size_t)MROWS * 4);
    int*    counts   = (int*)   carve(64);
    half_t* wf_h     = (half_t*)carve((size_t)MROWS * CCH * 2);
    float*  sw32     = (float*) carve((size_t)MROWS * TOK * 4);
    float*  pool32   = (float*) carve((size_t)KVROWS * CCH * 4);
    half_t* pool_h   = (half_t*)carve((size_t)KVROWS * CCH * 2);
    float*  q32      = (float*) carve((size_t)MROWS * CCH * 4);
    half_t* q_h      = (half_t*)carve((size_t)MROWS * CCH * 2);
    half_t* qq_h     = (half_t*)carve((size_t)MROWS * CCH * 2);
    half_t* kk_h     = (half_t*)carve((size_t)KVROWS * CCH * 2);
    half_t* vv_h     = (half_t*)carve((size_t)KVROWS * CCH * 2);
    half_t* o_h      = (half_t*)carve((size_t)MROWS * CCH * 2);
    float*  y32      = (float*) carve((size_t)MROWS * CCH * 4);
    half_t* h_h      = (half_t*)carve((size_t)MROWS * FFD * 2);

    auto cvt = [&](const float* s, half_t* d, int n) {
        cvt_f32_f16<<<(n + 255) / 256, 256, 0, stream>>>(s, d, n);
    };
    // weight f32 -> f16 (once per call; cheap)
    cvt(proj_w, w_proj_h, CCH * 512);
    cvt(soft_w, w_soft_h, 2 * TOK * CCH);
    cvt(ain_w,  w_ain_h,  2 * NLAY * 3 * CCH * CCH);
    cvt(aout_w, w_aout_h, 2 * NLAY * CCH * CCH);
    cvt(f1_w,   w_f1_h,   2 * NLAY * FFD * CCH);
    cvt(f2_w,   w_f2_h,   2 * NLAY * CCH * FFD);

    // concat+transpose -> f16 [B*N, 512]
    {
        dim3 g(NPIX / 32, 512 / 32, BSZ), b(32, 8);
        tconcat<<<g, b, 0, stream>>>(x, y, xcT_h);
    }
    // feat = xcT @ proj_w^T + proj_b
    gemm(stream, xcT_h, w_proj_h, proj_b, feat_t, feat_h, MROWS, CCH, 512, 0);

    // saliency + masks + counts
    zero_i32<<<1, 32, 0, stream>>>(counts, 16);
    sal_heat<<<MROWS, 256, 0, stream>>>(feat_t, sal_w, sal_b, heat, counts);

    // default output = fmap
    out_default<<<(BSZ * CCH * NPIX + 255) / 256, 256, 0, stream>>>(feat_t, out);

    const int EL = MROWS * CCH;
    for (int side = 0; side < 2; ++side) {
        wf_mul<<<(EL + 255) / 256, 256, 0, stream>>>(feat_t, heat, wf_h, side);
        // sw = wf @ soft_w^T + soft_b    [B*N, 64]
        gemm(stream, wf_h, w_soft_h + (size_t)side * TOK * CCH,
             soft_b + side * TOK, sw32, nullptr, MROWS, TOK, CCH, 0);
        // pool = sw^T * wf  [B*K, C]
        pool_red<<<KVROWS, 256, 0, stream>>>(sw32, wf_h, pool32, pool_h);
        // q <- q0
        q_init<<<(EL + 255) / 256, 256, 0, stream>>>(feat_t, q32, q_h);

        for (int l = 0; l < NLAY; ++l) {
            int sl = side * NLAY + l;
            const half_t* wq = w_ain_h + (size_t)sl * 3 * CCH * CCH;
            const half_t* wk = wq + (size_t)CCH * CCH;
            const half_t* wv = wq + (size_t)2 * CCH * CCH;
            const float*  bq = ain_b + (size_t)sl * 3 * CCH;
            const float*  bk = bq + CCH;
            const float*  bv = bq + 2 * CCH;

            gemm(stream, q_h,    wq, bq, nullptr, qq_h, MROWS,  CCH, CCH, 0);
            gemm(stream, pool_h, wk, bk, nullptr, kk_h, KVROWS, CCH, CCH, 0);
            gemm(stream, pool_h, wv, bv, nullptr, vv_h, KVROWS, CCH, CCH, 0);

            { dim3 g(NPIX / 128, BSZ * NHD); attn_wmma<<<g, 256, 0, stream>>>(qq_h, kk_h, vv_h, o_h); }

            gemm(stream, o_h, w_aout_h + (size_t)sl * CCH * CCH,
                 aout_b + (size_t)sl * CCH, y32, nullptr, MROWS, CCH, CCH, 0);
            ln_add<<<MROWS, 256, 0, stream>>>(q32, y32, n1_g + (size_t)sl * CCH,
                                              n1_b + (size_t)sl * CCH, q32, q_h);

            gemm(stream, q_h, w_f1_h + (size_t)sl * FFD * CCH,
                 f1_b + (size_t)sl * FFD, nullptr, h_h, MROWS, FFD, CCH, 1);
            gemm(stream, h_h, w_f2_h + (size_t)sl * CCH * FFD,
                 f2_b + (size_t)sl * CCH, y32, nullptr, MROWS, CCH, FFD, 0);
            ln_add<<<MROWS, 256, 0, stream>>>(q32, y32, n2_g + (size_t)sl * CCH,
                                              n2_b + (size_t)sl * CCH, q32, q_h);
        }
        scatter_side<<<MROWS, 256, 0, stream>>>(q32, heat, counts, out, side);
    }
}
